// GCN_69209103007771
// MI455X (gfx1250) — matrix operand order repaired and checked by reading, not compiled, not run on hardware
//
#include <hip/hip_runtime.h>
#include <math.h>

// ---------------------------------------------------------------------------
// CDNA5 / gfx1250 implementation of 3-stage SAG-pool GCN.
// GEMMs use v_wmma_f32_16x16x32_bf16 with split-precision (bf16 hi+lo) A/B.
// W is pre-split into packed bf16-pair arrays (column-transposed) once per
// GEMM; W chunks are staged into LDS with GLOBAL_LOAD_ASYNC_TO_LDS_B128
// (ASYNCcnt path) when the toolchain exposes the builtin.
// ---------------------------------------------------------------------------

typedef __attribute__((ext_vector_type(16))) __bf16       v16bf;
typedef __attribute__((ext_vector_type(8)))  float        v8f;
typedef __attribute__((ext_vector_type(8)))  unsigned int v8u;

#define HDIM 128
#define NPG  50      // nodes per graph
#define WSTR 20      // LDS row stride (uints) for W chunks: 80B = 16B-aligned

#if defined(__has_builtin)
#if __has_builtin(__builtin_amdgcn_global_load_async_to_lds_b128)
#define HAVE_ASYNC_LDS 1
#endif
#endif

#ifdef HAVE_ASYNC_LDS
typedef int v4i __attribute__((vector_size(16)));
typedef __attribute__((address_space(1))) v4i* gptr_b128;   // global int4*
typedef __attribute__((address_space(3))) v4i* lptr_b128;   // LDS int4*
__device__ __forceinline__ void wait_async() {
#if __has_builtin(__builtin_amdgcn_s_wait_asynccnt)
    __builtin_amdgcn_s_wait_asynccnt(0);
#else
    asm volatile("s_wait_asynccnt 0x0" ::: "memory");
#endif
}
#endif

// ---- bf16 split helpers (RTNE, pure bit math) ------------------------------
__device__ __forceinline__ unsigned f32_bf16_bits(float f) {
    unsigned u = __float_as_uint(f);
    u += 0x7FFFu + ((u >> 16) & 1u);      // round-to-nearest-even
    return u >> 16;
}
__device__ __forceinline__ float bf16_bits_f32(unsigned h) {
    return __uint_as_float(h << 16);
}
__device__ __forceinline__ void split_bits(float f, unsigned& hb, unsigned& lb) {
    hb = f32_bf16_bits(f);
    float r = f - bf16_bits_f32(hb);
    lb = f32_bf16_bits(r);
}

// ---- pre-split W (F x 128 fp32) into packed K-pair hi/lo uint arrays -------
// Column-transposed: WpHi[c*npairs + p] = hi(W[2p][c]) | hi(W[2p+1][c]) << 16
__global__ void k_splitw(const float* __restrict__ W,
                         unsigned* __restrict__ WpHi, unsigned* __restrict__ WpLo,
                         int npairs /* = F/2 */) {
    int idx = blockIdx.x * 256 + threadIdx.x;            // over npairs*128
    if (idx >= npairs * HDIM) return;
    int p = idx >> 7, c = idx & 127;
    float w0 = W[(size_t)(2 * p) * HDIM + c];
    float w1 = W[(size_t)(2 * p + 1) * HDIM + c];
    unsigned h0, l0, h1, l1;
    split_bits(w0, h0, l0);
    split_bits(w1, h1, l1);
    WpHi[(size_t)c * npairs + p] = h0 | (h1 << 16);
    WpLo[(size_t)c * npairs + p] = l0 | (l1 << 16);
}

// ---- GEMM: OUT(N x 128) = X(N x F) @ W(F x 128) [+bias][relu] --------------
// block = 256 threads = 8 wave32; block computes 16 rows x 128 cols.
// wave w owns the 16x16 tile at columns [16w, 16w+16).
__global__ __launch_bounds__(256) void k_gemm_wmma(
    const float* __restrict__ X,
    const unsigned* __restrict__ WpHi, const unsigned* __restrict__ WpLo,
    const float* __restrict__ bias, float* __restrict__ OUT,
    int N, int F, int do_relu)
{
    __shared__ unsigned lAhi[16 * 17];    // [m][pair]  pair = k/2 (0..15)
    __shared__ unsigned lAlo[16 * 17];
    __shared__ unsigned lWhi[128 * WSTR]; // [c][pair]  (16B-aligned rows)
    __shared__ unsigned lWlo[128 * WSTR];

    const int tid  = threadIdx.x;
    const int lane = tid & 31;
    const int wave = tid >> 5;
    const int half = lane >> 4;           // 0 or 1
    const int m    = lane & 15;
    const int m0   = blockIdx.x * 16;
    const int ncol = wave * 16 + m;       // output column of C fragment
    const int wstride = F >> 1;           // pairs per column in Wp arrays

    v8f acc = {};
    for (int kc = 0; kc < F; kc += 32) {
        const int pb = kc >> 1;           // global pair base for this chunk

        // ---- stage W chunk: 16 pairs x 128 cols
#ifdef HAVE_ASYNC_LDS
        // async DMA: 512 B128 transfers per array (2 per thread)
        #pragma unroll
        for (int i = 0; i < 2; ++i) {
            int idx = tid + i * 256;      // 0..511
            int c  = idx >> 2;
            int pg = (idx & 3) * 4;       // pair group of 4 (16B)
            __builtin_amdgcn_global_load_async_to_lds_b128(
                (gptr_b128)(WpHi + (size_t)c * wstride + pb + pg),
                (lptr_b128)&lWhi[c * WSTR + pg], 0, 0);
            __builtin_amdgcn_global_load_async_to_lds_b128(
                (gptr_b128)(WpLo + (size_t)c * wstride + pb + pg),
                (lptr_b128)&lWlo[c * WSTR + pg], 0, 0);
        }
#else
        #pragma unroll
        for (int i = 0; i < 8; ++i) {     // 2048 uints per array
            int idx = tid + i * 256;
            int p = idx & 15, c = idx >> 4;
            lWhi[c * WSTR + p] = WpHi[(size_t)c * wstride + pb + p];
            lWlo[c * WSTR + p] = WpLo[(size_t)c * wstride + pb + p];
        }
#endif
        // ---- stage A tile: 16x32 fp32 -> packed bf16 pairs (2 elems/thread)
        {
            int idx = tid * 2;            // even element index in 16x32 tile
            int r = idx >> 5, c0 = idx & 31;
            int gr = m0 + r;
            float v0 = (gr < N) ? X[(size_t)gr * F + kc + c0]     : 0.f;
            float v1 = (gr < N) ? X[(size_t)gr * F + kc + c0 + 1] : 0.f;
            unsigned h0, l0, h1, l1;
            split_bits(v0, h0, l0);
            split_bits(v1, h1, l1);
            lAhi[r * 17 + (c0 >> 1)] = h0 | (h1 << 16);
            lAlo[r * 17 + (c0 >> 1)] = l0 | (l1 << 16);
        }
#ifdef HAVE_ASYNC_LDS
        wait_async();                     // this wave's async LDS writes done
#endif
        __syncthreads();                  // publish LDS to all waves

        // ---- assemble fragments (pure ds_load + bit_cast)
        v8u ah, al, bh, bl;
        #pragma unroll
        for (int v = 0; v < 8; ++v) {
            int pa  = (v >> 2) * 8 + half * 4 + (v & 3); // A pair index
            int pbp = (v >> 2) * 8 + half * 4 + (v & 3); // B pair index
            ah[v] = lAhi[m * 17 + pa];
            al[v] = lAlo[m * 17 + pa];
            bh[v] = lWhi[ncol * WSTR + pbp];
            bl[v] = lWlo[ncol * WSTR + pbp];
        }
        v16bf a_hi = __builtin_bit_cast(v16bf, ah);
        v16bf a_lo = __builtin_bit_cast(v16bf, al);
        v16bf b_hi = __builtin_bit_cast(v16bf, bh);
        v16bf b_lo = __builtin_bit_cast(v16bf, bl);

        // split-precision accumulate: hi*hi + hi*lo + lo*hi
        acc = __builtin_amdgcn_wmma_f32_16x16x32_bf16(false, a_hi, false, b_hi,
                                                      (short)0, acc, false, false);
        acc = __builtin_amdgcn_wmma_f32_16x16x32_bf16(false, a_hi, false, b_lo,
                                                      (short)0, acc, false, false);
        acc = __builtin_amdgcn_wmma_f32_16x16x32_bf16(false, a_lo, false, b_hi,
                                                      (short)0, acc, false, false);
        __syncthreads();                  // protect LDS before next overwrite
    }

    float bcol = bias ? bias[ncol] : 0.f;
    #pragma unroll
    for (int r = 0; r < 8; ++r) {
        int row = m0 + r + half * 8;
        if (row < N) {
            float v = acc[r] + bcol;
            if (do_relu) v = fmaxf(v, 0.f);
            OUT[(size_t)row * HDIM + ncol] = v;
        }
    }
}

// ---- elementwise / edge kernels --------------------------------------------
__global__ void k_fill(float* __restrict__ p, float v, size_t n) {
    size_t i = (size_t)blockIdx.x * 256 + threadIdx.x;
    if (i < n) p[i] = v;
}

__global__ void k_edge_deg(const int* __restrict__ dst,
                           const float* __restrict__ emask,
                           float* __restrict__ deg, int E) {
    int e = blockIdx.x * 256 + threadIdx.x;
    if (e < E) {
        float w = emask[e];
        if (w != 0.f) atomicAdd(&deg[dst[e]], w);
    }
}

__global__ void k_dinv(const float* __restrict__ deg,
                       const float* __restrict__ nmask,
                       float* __restrict__ dinv, int N) {
    int n = blockIdx.x * 256 + threadIdx.x;
    if (n < N) {
        float d = deg[n] + nmask[n];
        dinv[n] = (d > 0.f) ? rsqrtf(fmaxf(d, 1e-12f)) : 0.f;
    }
}

// one wave per edge; lane l handles features l, l+32, l+64, l+96
__global__ __launch_bounds__(256) void k_edge_scatter(
    const float* __restrict__ feat, const int* __restrict__ src,
    const int* __restrict__ dst, const float* __restrict__ emask,
    const float* __restrict__ dinv, float* __restrict__ out,
    int E, int use_norm)
{
    int e = blockIdx.x * 8 + (threadIdx.x >> 5);
    if (e >= E) return;
    float w = emask[e];
    if (w == 0.f) return;
    int s = src[e], d = dst[e];
    int lane = threadIdx.x & 31;
    // warm the gathered row into cache early (global_prefetch_b8)
    __builtin_prefetch(feat + (size_t)s * HDIM + lane * 4, 0, 0);
    if (use_norm) {
        w *= dinv[s] * dinv[d];
        if (w == 0.f) return;
    }
    #pragma unroll
    for (int j = 0; j < 4; ++j) {
        int f = lane + 32 * j;
        atomicAdd(&out[(size_t)d * HDIM + f], feat[(size_t)s * HDIM + f] * w);
    }
}

// agg = agg + h * (dinv^2 * nmask) + cB   (in place on agg)
__global__ void k_gcn_combine(float* __restrict__ agg, const float* __restrict__ h,
                              const float* __restrict__ dinv,
                              const float* __restrict__ nmask,
                              const float* __restrict__ cB, size_t total) {
    size_t i = (size_t)blockIdx.x * 256 + threadIdx.x;
    if (i < total) {
        int n = (int)(i >> 7), f = (int)(i & 127);
        float di = dinv[n];
        agg[i] = agg[i] + h[i] * (di * di * nmask[n]) + cB[f];
    }
}

// stats[0..127]=sum(x*m), stats[128..255]=sum(x^2*m), stats[256]=sum(m)
__global__ __launch_bounds__(256) void k_bn_stats(
    const float* __restrict__ x, const float* __restrict__ nmask,
    float* __restrict__ stats, int N)
{
    __shared__ float sh[2 * HDIM];
    int tid = threadIdx.x;
    int f = tid & 127, r = tid >> 7;
    int n0 = blockIdx.x * 128;
    float s = 0.f, sq = 0.f, cm = 0.f;
    for (int i = r; i < 128; i += 2) {
        int n = n0 + i;
        if (n < N) {
            float mval = nmask[n];
            float xv = x[(size_t)n * HDIM + f];
            float v = xv * mval;
            s += v;
            sq += v * xv;          // = x^2 * m for m in {0,1}
            if (f == 0) cm += mval;
        }
    }
    if (r == 1) { sh[f] = s; sh[HDIM + f] = sq; }
    __syncthreads();
    if (r == 0) {
        s += sh[f]; sq += sh[HDIM + f];
        atomicAdd(&stats[f], s);
        atomicAdd(&stats[HDIM + f], sq);
    }
    if (f == 0) atomicAdd(&stats[2 * HDIM], cm);
}

__global__ void k_bn_apply(const float* __restrict__ x,
                           const float* __restrict__ stats,
                           const float* __restrict__ g, const float* __restrict__ bt,
                           const float* __restrict__ nmask,
                           float* __restrict__ out, size_t total) {
    size_t i = (size_t)blockIdx.x * 256 + threadIdx.x;
    if (i >= total) return;
    int n = (int)(i >> 7), f = (int)(i & 127);
    float cnt  = fmaxf(stats[2 * HDIM], 1.f);
    float mean = stats[f] / cnt;
    float var  = fmaxf(stats[HDIM + f] / cnt - mean * mean, 0.f);
    float val  = g[f] * (x[i] - mean) * rsqrtf(var + 1e-5f) + bt[f];
    out[i] = val * nmask[n];
}

// score = tanh(agg2@Wr + Br + x@Wro) ; one wave per node
__global__ __launch_bounds__(256) void k_score(
    const float* __restrict__ agg2, const float* __restrict__ xbn,
    const float* __restrict__ Wr, const float* __restrict__ Br,
    const float* __restrict__ Wro, float* __restrict__ sarr, int N)
{
    int node = blockIdx.x * 8 + (threadIdx.x >> 5);
    if (node >= N) return;
    int lane = threadIdx.x & 31;
    float p = 0.f;
    #pragma unroll
    for (int j = 0; j < 4; ++j) {
        int f = lane + 32 * j;
        p += agg2[(size_t)node * HDIM + f] * Wr[f]
           + xbn[(size_t)node * HDIM + f] * Wro[f];
    }
    #pragma unroll
    for (int off = 16; off; off >>= 1) p += __shfl_xor(p, off, 32);
    if (lane == 0) sarr[node] = tanhf(p + Br[0]);
}

// per-graph top-k (ceil(0.5*cnt)) by score; matches reference lexsort semantics
__global__ __launch_bounds__(64) void k_topk(const float* __restrict__ sarr,
                                             float* __restrict__ nmask, int G) {
    __shared__ float ss[NPG], mm[NPG];
    int g = blockIdx.x, t = threadIdx.x;
    if (t < NPG) { ss[t] = sarr[g * NPG + t]; mm[t] = nmask[g * NPG + t]; }
    __syncthreads();
    if (t >= NPG) return;
    float cnt = 0.f;
    for (int j = 0; j < NPG; ++j) cnt += mm[j];
    float ki = (mm[t] > 0.f) ? ss[t] : -__builtin_inff();
    int rank = 0;
    for (int j = 0; j < NPG; ++j) {
        float kj = (mm[j] > 0.f) ? ss[j] : -__builtin_inff();
        if (kj > ki || (kj == ki && j < t)) rank++;
    }
    float k = ceilf(0.5f * cnt);
    float keep = ((float)rank < k) ? 1.f : 0.f;
    nmask[g * NPG + t] = keep * mm[t];
}

__global__ void k_xnew(float* __restrict__ x, const float* __restrict__ sarr,
                       const float* __restrict__ nmask, size_t total) {
    size_t i = (size_t)blockIdx.x * 256 + threadIdx.x;
    if (i < total) {
        int n = (int)(i >> 7);
        x[i] *= sarr[n] * nmask[n];
    }
}

__global__ void k_emask(float* __restrict__ emask, const int* __restrict__ src,
                        const int* __restrict__ dst,
                        const float* __restrict__ nmask, int E) {
    int e = blockIdx.x * 256 + threadIdx.x;
    if (e < E) {
        float v = emask[e];
        if (v != 0.f) emask[e] = v * nmask[src[e]] * nmask[dst[e]];
    }
}

// mean pool per graph; pruned nodes already zero in x
__global__ __launch_bounds__(128) void k_pool(const float* __restrict__ x,
                                              const float* __restrict__ nmask,
                                              float* __restrict__ out, int G) {
    __shared__ float mm[NPG];
    int g = blockIdx.x, t = threadIdx.x;
    if (t < NPG) mm[t] = nmask[g * NPG + t];
    __syncthreads();
    float cnt = 0.f, acc = 0.f;
    for (int i = 0; i < NPG; ++i) {
        cnt += mm[i];
        acc += x[(size_t)(g * NPG + i) * HDIM + t];
    }
    out[(size_t)g * HDIM + t] = acc / fmaxf(cnt, 1.f);
}

// ---------------------------------------------------------------------------
struct StageP { const float *cW, *cB, *fW, *fB, *g, *bt, *Wr, *Br, *Wro; };

static StageP unpack_flat(void* const* d, int base) {
    StageP p;
    p.cW = (const float*)d[base + 0]; p.cB = (const float*)d[base + 1];
    p.fW = (const float*)d[base + 2]; p.fB = (const float*)d[base + 3];
    p.g  = (const float*)d[base + 4]; p.bt = (const float*)d[base + 5];
    p.Wr = (const float*)d[base + 6]; p.Br = (const float*)d[base + 7];
    p.Wro = (const float*)d[base + 8];
    return p;
}
static StageP unpack_packed(const float* buf, int fin) {
    StageP p; size_t o = 0;
    p.cW = buf + o; o += (size_t)fin * HDIM;
    p.cB = buf + o; o += HDIM;
    p.fW = buf + o; o += (size_t)HDIM * HDIM;
    p.fB = buf + o; o += HDIM;
    p.g  = buf + o; o += HDIM;
    p.bt = buf + o; o += HDIM;
    p.Wr = buf + o; o += HDIM;
    p.Br = buf + o; o += 1;
    p.Wro = buf + o;
    return p;
}

extern "C" void kernel_launch(void* const* d_in, const int* in_sizes, int n_in,
                              void* d_out, int out_size, void* d_ws, size_t ws_size,
                              hipStream_t stream)
{
    const float* x  = (const float*)d_in[0];
    const int*   ei = (const int*)d_in[1];
    const int N   = in_sizes[2];
    const int E   = in_sizes[1] / 2;
    const int FIN = in_sizes[0] / N;
    const int G   = out_size / HDIM;
    const int* src = ei;
    const int* dst = ei + E;

    StageP P[3];
    if (n_in >= 30) {
        P[0] = unpack_flat(d_in, 3);
        P[1] = unpack_flat(d_in, 12);
        P[2] = unpack_flat(d_in, 21);
    } else {
        P[0] = unpack_packed((const float*)d_in[3], FIN);
        P[1] = unpack_packed((const float*)d_in[4], HDIM);
        P[2] = unpack_packed((const float*)d_in[5], HDIM);
    }

    // workspace carve (~111 MB total)
    char* base = (char*)d_ws;
    size_t off = 0;
    auto carve = [&](size_t nbytes) -> void* {
        void* p = (void*)(base + off);
        off += ((nbytes + 255) / 256) * 256;
        return p;
    };
    float* B1    = (float*)carve((size_t)N * HDIM * 4);
    float* B2    = (float*)carve((size_t)N * HDIM * 4);
    float* deg   = (float*)carve((size_t)N * 4);
    float* dinv  = (float*)carve((size_t)N * 4);
    float* sarr  = (float*)carve((size_t)N * 4);
    float* nmask = (float*)carve((size_t)N * 4);
    float* emask = (float*)carve((size_t)E * 4);
    float* stats = (float*)carve((size_t)(2 * HDIM + 8) * 4);
    unsigned* WpHi = (unsigned*)carve((size_t)(FIN / 2) * HDIM * 4);
    unsigned* WpLo = (unsigned*)carve((size_t)(FIN / 2) * HDIM * 4);
    (void)ws_size;

    const size_t totNF = (size_t)N * HDIM;
    const int blkNF   = (int)((totNF + 255) / 256);
    const int blkN    = (N + 255) / 256;
    const int blkE    = (E + 255) / 256;
    const int blkEw   = (E + 7) / 8;       // wave per edge
    const int blkWv   = (N + 7) / 8;       // wave per node
    const int blkRows = (N + 15) / 16;
    const int blkBN   = (N + 127) / 128;

    k_fill<<<blkN, 256, 0, stream>>>(nmask, 1.f, (size_t)N);
    k_fill<<<blkE, 256, 0, stream>>>(emask, 1.f, (size_t)E);

    const float* xin = x;
    int F = FIN;
    for (int s = 0; s < 3; ++s) {
        const StageP& p = P[s];
        // h = xin @ cW -> B1
        {
            int npairs = F / 2;
            int blkW = (npairs * HDIM + 255) / 256;
            k_splitw<<<blkW, 256, 0, stream>>>(p.cW, WpHi, WpLo, npairs);
            k_gemm_wmma<<<blkRows, 256, 0, stream>>>(xin, WpHi, WpLo, nullptr, B1, N, F, 0);
        }
        // degree / dinv
        k_fill<<<blkN, 256, 0, stream>>>(deg, 0.f, (size_t)N);
        k_edge_deg<<<blkE, 256, 0, stream>>>(dst, emask, deg, E);
        k_dinv<<<blkN, 256, 0, stream>>>(deg, nmask, dinv, N);
        // agg -> B2 ; combine (+self term +bias) in place
        k_fill<<<blkNF, 256, 0, stream>>>(B2, 0.f, totNF);
        k_edge_scatter<<<blkEw, 256, 0, stream>>>(B1, src, dst, emask, dinv, B2, E, 1);
        k_gcn_combine<<<blkNF, 256, 0, stream>>>(B2, B1, dinv, nmask, p.cB, totNF);
        // h2 = relu(B2 @ fW + fB) -> B1
        {
            int npairs = HDIM / 2;
            int blkW = (npairs * HDIM + 255) / 256;
            k_splitw<<<blkW, 256, 0, stream>>>(p.fW, WpHi, WpLo, npairs);
            k_gemm_wmma<<<blkRows, 256, 0, stream>>>(B2, WpHi, WpLo, p.fB, B1, N, HDIM, 1);
        }
        // masked BN -> B2
        k_fill<<<1, 256, 0, stream>>>(stats, 0.f, (size_t)(2 * HDIM + 8));
        k_bn_stats<<<blkBN, 256, 0, stream>>>(B1, nmask, stats, N);
        k_bn_apply<<<blkNF, 256, 0, stream>>>(B1, stats, p.g, p.bt, nmask, B2, totNF);
        // SAG pool: agg2 -> B1 ; score ; top-k ; apply
        k_fill<<<blkNF, 256, 0, stream>>>(B1, 0.f, totNF);
        k_edge_scatter<<<blkEw, 256, 0, stream>>>(B2, src, dst, emask, nullptr, B1, E, 0);
        k_score<<<blkWv, 256, 0, stream>>>(B1, B2, p.Wr, p.Br, p.Wro, sarr, N);
        k_topk<<<G, 64, 0, stream>>>(sarr, nmask, G);
        k_xnew<<<blkNF, 256, 0, stream>>>(B2, sarr, nmask, totNF);
        k_emask<<<blkE, 256, 0, stream>>>(emask, src, dst, nmask, E);
        xin = B2; F = HDIM;
    }
    k_pool<<<G, 128, 0, stream>>>(B2, nmask, (float*)d_out, G);
}
